// Social_SGNet_CVAE_85109071938035
// MI455X (gfx1250) — matrix operand order repaired and checked by reading, not compiled, not run on hardware
//
#include <hip/hip_runtime.h>
#include <hip/hip_bf16.h>
#include <math.h>

// ---------------------------------------------------------------------------
// Social_SGNet_CVAE forward for MI455X (gfx1250, wave32, WMMA bf16).
// All dense layers run through a double-buffered 128x64-tile WMMA GEMM
// (v_wmma_f32_16x16x32_bf16, f32 accumulate). Weight tiles staged to LDS via
// the gfx1250 async global->LDS path (ASYNCcnt) when the toolchain exposes it.
// ---------------------------------------------------------------------------

typedef __bf16 bf16;
typedef bf16 v16bf __attribute__((ext_vector_type(16)));
typedef bf16 v8bf  __attribute__((ext_vector_type(8)));
typedef float v8f  __attribute__((ext_vector_type(8)));
typedef int  i32x4 __attribute__((__vector_size__(16)));

#define NAG   256   // N agents
#define TENC  8
#define TDEC  12
#define KS    20
#define HDIM  512
#define HQ    128
#define LAT   32
#define FEAT  256
#define NEMB  128
#define NK    (NAG*KS)   // 5120

// gfx1250 async global->LDS copy (ASYNCcnt-tracked), probed via __has_builtin.
#if defined(__HIP_DEVICE_COMPILE__) && \
    __has_builtin(__builtin_amdgcn_global_load_async_to_lds_b128) && \
    __has_builtin(__builtin_amdgcn_s_wait_asynccnt)
#define USE_ASYNC 1
#define AS1 __attribute__((address_space(1)))
#define AS3 __attribute__((address_space(3)))
#else
#define USE_ASYNC 0
#endif

static __device__ __forceinline__ void wait_async_all() {
#if USE_ASYNC
  __builtin_amdgcn_s_wait_asynccnt(0);
#endif
}

static __device__ __forceinline__ float sigmoidf_(float x) {
  return 1.f / (1.f + __expf(-x));
}

// ------------------------------ WMMA GEMM ----------------------------------
// C[M][Nout] = act( A[M][Kin] * W^T + bias ),  W stored bf16 [Nout][Kin].
// Requires Kin % 32 == 0 (all call sites: 128/256/512/544/640/768).
// Out-of-range A rows / W rows are index-CLAMPED (never branched): the
// products land only in C entries that are masked at the store.
template<int ABF, int CBF>
__global__ __launch_bounds__(256)
void gemm_wmma(const void* __restrict__ Ain, int lda,
               const bf16* __restrict__ W,
               const float* __restrict__ bias,
               void* __restrict__ Cout, int ldc,
               int M, int Nout, int Kin, int act)
{
  __shared__ __align__(32) bf16 sA[2][128][32];   // 2 x 8 KB
  __shared__ __align__(32) bf16 sB[2][64][32];    // 2 x 4 KB
  const int tid  = threadIdx.x;
  const int lane = tid & 31;
  const int wave = tid >> 5;                      // 0..7 -> 16-row strip
  const int m0 = blockIdx.x * 128;
  const int n0 = blockIdx.y * 64;

  const float* Af = (const float*)Ain;
  const bf16*  Ab = (const bf16*)Ain;

  v8f acc[4] = {};

  // ---- tile stager: B always bf16 (16B/thread); A bf16 (32B/thread) or
  //      fp32 with on-the-fly convert (64B/thread). Indices clamped.
  auto stage = [&](int buf, int k0) {
    {
      int n = tid >> 2, c = (tid & 3) * 8;
      int gn = min(n0 + n, Nout - 1);
      const bf16* gp = &W[(size_t)gn * Kin + k0 + c];
#if USE_ASYNC
      __builtin_amdgcn_global_load_async_to_lds_b128(
          (AS1 i32x4*)gp, (AS3 i32x4*)&sB[buf][n][c], 0, 0);
#else
      *(float4*)&sB[buf][n][c] = *(const float4*)gp;
#endif
    }
    if (ABF) {
      #pragma unroll
      for (int it = 0; it < 2; ++it) {
        int q = tid + it * 256;            // 512 chunks of 16B
        int r = q >> 2, c = (q & 3) * 8;
        int gm = min(m0 + r, M - 1);
        const bf16* gp = &Ab[(size_t)gm * lda + k0 + c];
#if USE_ASYNC
        __builtin_amdgcn_global_load_async_to_lds_b128(
            (AS1 i32x4*)gp, (AS3 i32x4*)&sA[buf][r][c], 0, 0);
#else
        *(float4*)&sA[buf][r][c] = *(const float4*)gp;
#endif
      }
    } else {
      int r = tid >> 1, cb = (tid & 1) * 16;   // 16 floats per thread
      int gm = min(m0 + r, M - 1);
      const float* gp = &Af[(size_t)gm * lda + k0 + cb];
      #pragma unroll
      for (int v = 0; v < 4; ++v) {
        float4 f = *(const float4*)(gp + v * 4);
        int c = cb + v * 4;
        sA[buf][r][c + 0] = (bf16)f.x; sA[buf][r][c + 1] = (bf16)f.y;
        sA[buf][r][c + 2] = (bf16)f.z; sA[buf][r][c + 3] = (bf16)f.w;
      }
    }
  };

  const int nK = Kin >> 5;
  stage(0, 0);

  for (int kt = 0; kt < nK; ++kt) {
    const int buf = kt & 1;
    wait_async_all();          // this wave's async copies into `buf` done
    __syncthreads();           // everyone's copies into `buf` visible
    if (kt + 1 < nK) stage(buf ^ 1, (kt + 1) << 5);   // overlap with WMMA

    // ---- A fragment: 16x32 bf16; lane m = lane&15, 16 contiguous K at (lane>>4)*16 ----
    const int am  = (wave << 4) + (lane & 15);
    const int akh = (lane >> 4) << 4;
    v8bf lo = *(const v8bf*)&sA[buf][am][akh];
    v8bf hi = *(const v8bf*)&sA[buf][am][akh + 8];
    v16bf afrag = __builtin_shufflevector(lo, hi, 0, 1, 2, 3, 4, 5, 6, 7,
                                          8, 9, 10, 11, 12, 13, 14, 15);

    // ---- B fragments: col = lane&15, 16 contiguous K at (lane>>4)*16 ----
    const int bkh = (lane >> 4) << 4;
    #pragma unroll
    for (int nt = 0; nt < 4; ++nt) {
      const int bc = (nt << 4) + (lane & 15);
      v16bf bfrag = *(const v16bf*)&sB[buf][bc][bkh];
      acc[nt] = __builtin_amdgcn_wmma_f32_16x16x32_bf16(
          false, afrag, false, bfrag, (short)0, acc[nt], false, false);
    }
    // next iteration's leading barrier also fences re-staging of `buf`
  }

  // ---- epilogue: C[m][n], m = wave*16 + i + 8*(lane>>4), n = nt*16 + lane&15
  float* Cf = (float*)Cout;
  bf16*  Cb = (bf16*)Cout;
  #pragma unroll
  for (int nt = 0; nt < 4; ++nt) {
    int n = n0 + (nt << 4) + (lane & 15);
    if (n >= Nout) continue;
    float bv = bias ? bias[n] : 0.f;
    #pragma unroll
    for (int i = 0; i < 8; ++i) {
      int m = m0 + (wave << 4) + i + ((lane >> 4) << 3);
      if (m >= M) continue;
      float v = acc[nt][i] + bv;
      if (act == 1) v = fmaxf(v, 0.f);
      if (CBF) Cb[(size_t)m * ldc + n] = (bf16)v;
      else     Cf[(size_t)m * ldc + n] = v;
    }
  }
}

// ------------------------- small helper kernels ----------------------------
__global__ void k_wcvt(const float* __restrict__ src, bf16* __restrict__ dst,
                       int Nr, int Kc, int trans)
{
  int i = blockIdx.x * 256 + threadIdx.x;
  if (i >= Nr * Kc) return;
  int n = i / Kc, k = i % Kc;
  float v = trans ? src[(size_t)k * Nr + n] : src[i];
  dst[i] = (bf16)v;
}

__global__ void k_fill(float* __restrict__ p, int n, float v)
{
  int i = blockIdx.x * 256 + threadIdx.x;
  if (i < n) p[i] = v;
}

__global__ void k_copy(float* __restrict__ d, const float* __restrict__ s, int n)
{
  int i = blockIdx.x * 256 + threadIdx.x;
  if (i < n) d[i] = s[i];
}

// TTC per agent pair (with the ref's where/min/NaN chain folded in).
__global__ void k_social_pairs(const float* __restrict__ raw, int s,
                               float* __restrict__ ttc, float* __restrict__ dpv)
{
  int idx = blockIdx.x * 256 + threadIdx.x;
  if (idx >= NAG * NAG) return;
  int i = idx >> 8, j = idx & 255;
  const float* ri = raw + ((size_t)i * TENC + s) * 6;
  const float* rj = raw + ((size_t)j * TENC + s) * 6;
  float dpx = ri[0] - rj[0], dpy = ri[1] - rj[1];
  float dvx = ri[2] - rj[2], dvy = ri[3] - rj[3];
  float dot  = dpx * dvx + dpy * dvy;
  float dvn2 = dvx * dvx + dvy * dvy;
  float t;
  if (dvn2 == 0.f) t = (dot == 0.f) ? 0.f : 12.f;        // nan->0, +-inf->12
  else { t = -dot / dvn2; if (t < 0.f) t = 12.f; t = fminf(t, 12.f); }
  ttc[idx] = t;
  float* d = dpv + (size_t)idx * 4;
  d[0] = dpx; d[1] = dpy; d[2] = dvx; d[3] = dvy;
}

// sigma = std(ttc, ddof=1) over 65536 values (single block).
__global__ __launch_bounds__(256)
void k_stats(const float* __restrict__ ttc, float* __restrict__ stats)
{
  __shared__ float ss[256], sq[256];
  float s = 0.f, q = 0.f;
  for (int i = threadIdx.x; i < NAG * NAG; i += 256) {
    float v = ttc[i]; s += v; q += v * v;
  }
  ss[threadIdx.x] = s; sq[threadIdx.x] = q; __syncthreads();
  for (int off = 128; off > 0; off >>= 1) {
    if (threadIdx.x < off) {
      ss[threadIdx.x] += ss[threadIdx.x + off];
      sq[threadIdx.x] += sq[threadIdx.x + off];
    }
    __syncthreads();
  }
  if (threadIdx.x == 0) {
    float cnt = (float)(NAG * NAG);
    float mean = ss[0] / cnt;
    float var  = (sq[0] - cnt * mean * mean) / (cnt - 1.f);
    stats[0] = sqrtf(fmaxf(var, 0.f));
  }
}

// Fused: rbf = exp(-ttc^2/2s^2)*(i!=j); A1 = bf16(relu(rbf*W1+b1)) [65536x256];
//        NN = relu([dp,dv] @ Wn + bn) [65536x128].  One block per pair.
__global__ __launch_bounds__(256)
void k_social_embed(const float* __restrict__ ttc, const float* __restrict__ dpv,
                    const float* __restrict__ stats,
                    const float* __restrict__ W1, const float* __restrict__ b1,
                    const float* __restrict__ Wn, const float* __restrict__ bn,
                    bf16* __restrict__ A1, float* __restrict__ NN)
{
  int p = blockIdx.x;
  int i = p >> 8, j = p & 255;
  float sg = stats[0];
  float t = ttc[p];
  float rbf = __expf(-(t * t) / (2.f * sg * sg));
  if (i == j) rbf = 0.f;
  int f = threadIdx.x;
  A1[(size_t)p * FEAT + f] = (bf16)fmaxf(rbf * W1[f] + b1[f], 0.f);
  if (f < NEMB) {
    const float* d = dpv + (size_t)p * 4;
    float acc = bn[f];
    #pragma unroll
    for (int c = 0; c < 4; ++c) acc += d[c] * Wn[c * NEMB + f];
    NN[(size_t)p * NEMB + f] = fmaxf(acc, 0.f);
  }
}

// e = leaky_relu(k . q), softmax over m, sf = att @ nn_; writes sw*sf into CI.
__global__ __launch_bounds__(256)
void k_social_attn(const bf16* __restrict__ KM, const float* __restrict__ Q,
                   const float* __restrict__ NN, const float* __restrict__ sw,
                   float* __restrict__ CI)
{
  int n = blockIdx.x, tid = threadIdx.x;
  __shared__ float e[256], red[256];
  const float* q  = Q + (size_t)n * FEAT;
  const bf16*  kr = KM + ((size_t)n * NAG + tid) * FEAT;
  float acc = 0.f;
  for (int d = 0; d < FEAT; ++d) acc += (float)kr[d] * q[d];
  acc = (acc > 0.f) ? acc : 0.2f * acc;
  e[tid] = acc; red[tid] = acc; __syncthreads();
  for (int off = 128; off > 0; off >>= 1) {
    if (tid < off) red[tid] = fmaxf(red[tid], red[tid + off]);
    __syncthreads();
  }
  float mx = red[0]; __syncthreads();
  float ex = __expf(e[tid] - mx);
  e[tid] = ex; red[tid] = ex; __syncthreads();
  for (int off = 128; off > 0; off >>= 1) {
    if (tid < off) red[tid] += red[tid + off];
    __syncthreads();
  }
  float inv = 1.f / red[0];
  if (tid < NEMB) {
    float sacc = 0.f;
    for (int m = 0; m < NAG; ++m)
      sacc += e[m] * inv * NN[((size_t)n * NAG + m) * NEMB + tid];
    CI[(size_t)n * 768 + 512 + tid] = sw[0] * sacc;
  }
}

// CI[:,0:512] = traj_input[:,s,:]; CI[:,640:768] = goal_for_enc.
__global__ void k_build_ci(const float* __restrict__ traj, int s,
                           const float* __restrict__ GFE, float* __restrict__ CI)
{
  int idx = blockIdx.x * 256 + threadIdx.x;
  if (idx >= NAG * 640) return;
  int n = idx / 640, c = idx % 640;
  if (c < 512) CI[(size_t)n * 768 + c] = traj[((size_t)n * TENC + s) * HDIM + c];
  else         CI[(size_t)n * 768 + 128 + c] = GFE[(size_t)n * HQ + (c - 512)];
}

// GRU gate fusion (biases already added inside gi/gh GEMMs). In-place on h.
__global__ void k_gru_gate(const float* __restrict__ gi, const float* __restrict__ gh,
                           float* __restrict__ h, int M, int dh)
{
  int idx = blockIdx.x * 256 + threadIdx.x;
  if (idx >= M * dh) return;
  int r = idx / dh, c = idx % dh;
  const float* gir = gi + (size_t)r * 3 * dh;
  const float* ghr = gh + (size_t)r * 3 * dh;
  float rg = sigmoidf_(gir[c] + ghr[c]);
  float z  = sigmoidf_(gir[dh + c] + ghr[dh + c]);
  float nn = tanhf(gir[2 * dh + c] + rg * ghr[2 * dh + c]);
  h[idx] = (1.f - z) * nn + z * h[idx];
}

// goal_for_enc: a = relu(tanh(gfe) . wa + ba), softmax over TDEC, weighted sum.
__global__ __launch_bounds__(128)
void k_enc_attn(const float* __restrict__ GFES,     // [TDEC][NAG][HQ]
                const float* __restrict__ wa, const float* __restrict__ ba,
                float* __restrict__ GFE)            // [NAG][HQ]
{
  int n = blockIdx.x, tid = threadIdx.x;
  __shared__ float a[TDEC];
  if (tid < TDEC) {
    const float* g = GFES + ((size_t)tid * NAG + n) * HQ;
    float acc = ba[0];
    for (int d = 0; d < HQ; ++d) acc += tanhf(g[d]) * wa[d];
    a[tid] = fmaxf(acc, 0.f);
  }
  __syncthreads();
  if (tid == 0) {
    float mx = a[0];
    for (int t = 1; t < TDEC; ++t) mx = fmaxf(mx, a[t]);
    float s = 0.f;
    for (int t = 0; t < TDEC; ++t) { a[t] = __expf(a[t] - mx); s += a[t]; }
    float inv = 1.f / s;
    for (int t = 0; t < TDEC; ++t) a[t] *= inv;
  }
  __syncthreads();
  float acc = 0.f;
  for (int t = 0; t < TDEC; ++t) acc += a[t] * GFES[((size_t)t * NAG + n) * HQ + tid];
  GFE[(size_t)n * HQ + tid] = acc;
}

// Decoder attention for step tcur; masked rows contribute relu(ba) to softmax
// but zero to gvec. Broadcasts gvec into XDEC[:,0:128] for all K samples.
__global__ __launch_bounds__(128)
void k_dec_attn(const float* __restrict__ GFD,      // [TDEC][NAG][HQ]
                const float* __restrict__ wa, const float* __restrict__ ba,
                int tcur, float* __restrict__ XDEC) // [NK][640]
{
  int n = blockIdx.x, tid = threadIdx.x;
  __shared__ float a[TDEC];
  if (tid < TDEC) {
    float acc = ba[0];
    if (tid >= tcur) {
      const float* g = GFD + ((size_t)tid * NAG + n) * HQ;
      for (int d = 0; d < HQ; ++d) acc += tanhf(g[d]) * wa[d];
    }
    a[tid] = fmaxf(acc, 0.f);
  }
  __syncthreads();
  if (tid == 0) {
    float mx = a[0];
    for (int t = 1; t < TDEC; ++t) mx = fmaxf(mx, a[t]);
    float s = 0.f;
    for (int t = 0; t < TDEC; ++t) { a[t] = __expf(a[t] - mx); s += a[t]; }
    float inv = 1.f / s;
    for (int t = 0; t < TDEC; ++t) a[t] *= inv;
  }
  __syncthreads();
  float acc = 0.f;
  for (int t = tcur; t < TDEC; ++t) acc += a[t] * GFD[((size_t)t * NAG + n) * HQ + tid];
  for (int k = 0; k < KS; ++k)
    XDEC[((size_t)(n * KS + k)) * 640 + tid] = acc;
}

// cvae_h = [broadcast(dec_h0), mu + exp(0.5*logvar)*eps]  -> [NK][544]
__global__ void k_build_xcvae(const float* __restrict__ DEC0, const float* __restrict__ MU,
                              const float* __restrict__ LV, const float* __restrict__ eps,
                              int s, float* __restrict__ X)
{
  int idx = blockIdx.x * 256 + threadIdx.x;
  if (idx >= NK * 544) return;
  int r = idx / 544, c = idx % 544;
  int n = r / KS, k = r % KS;
  float v;
  if (c < HDIM) v = DEC0[(size_t)n * HDIM + c];
  else {
    int l = c - HDIM;
    v = MU[(size_t)n * LAT + l] +
        __expf(0.5f * LV[(size_t)n * LAT + l]) *
        eps[(((size_t)n * TENC + s) * KS + k) * LAT + l];
  }
  X[idx] = v;
}

// regressor (dout=2) on goal path: T is [TDEC][NAG][512] -> out0[n][s][t][2]
__global__ void k_reg_goal(const float* __restrict__ T,
                           const float* __restrict__ W, const float* __restrict__ b,
                           int s, float* __restrict__ out0)
{
  int idx = blockIdx.x * 256 + threadIdx.x;
  if (idx >= TDEC * NAG * 2) return;
  int p = idx >> 1, c = idx & 1;
  int t = p / NAG, n = p % NAG;
  const float* row = T + (size_t)p * HDIM;
  float acc = b[c];
  for (int d = 0; d < HDIM; ++d) acc += row[d] * W[d * 2 + c];
  out0[(((size_t)n * TENC + s) * TDEC + t) * 2 + c] = acc;
}

// regressor on decoder hidden: DH [NK][512] -> out1[n][s][t][k][2]
__global__ void k_reg_dec(const float* __restrict__ DH,
                          const float* __restrict__ W, const float* __restrict__ b,
                          int s, int t, float* __restrict__ out1)
{
  int idx = blockIdx.x * 256 + threadIdx.x;
  if (idx >= NK * 2) return;
  int p = idx >> 1, c = idx & 1;
  int n = p / KS, k = p % KS;
  const float* row = DH + (size_t)p * HDIM;
  float acc = b[c];
  for (int d = 0; d < HDIM; ++d) acc += row[d] * W[d * 2 + c];
  out1[((((size_t)n * TENC + s) * TDEC + t) * KS + k) * 2 + c] = acc;
}

// ------------------------------- launcher ----------------------------------
extern "C" void kernel_launch(void* const* d_in, const int* in_sizes, int n_in,
                              void* d_out, int out_size, void* d_ws, size_t ws_size,
                              hipStream_t stream)
{
  (void)in_sizes; (void)n_in; (void)out_size; (void)ws_size;
  const float* raw   = (const float*)d_in[0];
  const float* traj  = (const float*)d_in[1];
  const float* eps   = (const float*)d_in[2];
  const float* sw    = (const float*)d_in[3];
  const float* W_k1  = (const float*)d_in[4];  const float* b_k1  = (const float*)d_in[5];
  const float* W_k2  = (const float*)d_in[6];  const float* b_k2  = (const float*)d_in[7];
  const float* W_nn  = (const float*)d_in[8];  const float* b_nn  = (const float*)d_in[9];
  const float* W_q   = (const float*)d_in[10]; const float* b_q   = (const float*)d_in[11];
  const float* Wih_t = (const float*)d_in[12]; const float* Whh_t = (const float*)d_in[13];
  const float* bih_t = (const float*)d_in[14]; const float* bhh_t = (const float*)d_in[15];
  const float* W_e2g = (const float*)d_in[16]; const float* b_e2g = (const float*)d_in[17];
  const float* Wih_g = (const float*)d_in[18]; const float* Whh_g = (const float*)d_in[19];
  const float* bih_g = (const float*)d_in[20]; const float* bhh_g = (const float*)d_in[21];
  const float* W_g2t = (const float*)d_in[22]; const float* b_g2t = (const float*)d_in[23];
  const float* W_reg = (const float*)d_in[24]; const float* b_reg = (const float*)d_in[25];
  const float* W_g2e = (const float*)d_in[26]; const float* b_g2e = (const float*)d_in[27];
  const float* W_g2d = (const float*)d_in[28]; const float* b_g2d = (const float*)d_in[29];
  const float* W_ea  = (const float*)d_in[30]; const float* b_ea  = (const float*)d_in[31];
  const float* W_da  = (const float*)d_in[32]; const float* b_da  = (const float*)d_in[33];
  const float* W_e2d = (const float*)d_in[34]; const float* b_e2d = (const float*)d_in[35];
  const float* W_pm  = (const float*)d_in[36]; const float* b_pm  = (const float*)d_in[37];
  const float* W_pl  = (const float*)d_in[38]; const float* b_pl  = (const float*)d_in[39];
  const float* W_c2d = (const float*)d_in[40]; const float* b_c2d = (const float*)d_in[41];
  const float* W_d2i = (const float*)d_in[42]; const float* b_d2i = (const float*)d_in[43];
  const float* Wih_d = (const float*)d_in[44]; const float* Whh_d = (const float*)d_in[45];
  const float* bih_d = (const float*)d_in[46]; const float* bhh_d = (const float*)d_in[47];

  float* out0 = (float*)d_out;                       // (N,T_ENC,T_DEC,2)
  float* out1 = out0 + (size_t)NAG * TENC * TDEC * 2;          // (N,TE,TD,K,2)
  float* out2 = out1 + (size_t)NAG * TENC * TDEC * KS * 2;     // (N,TE,K)

  // ---- workspace bump allocator (256B aligned) ----
  char* base = (char*)d_ws; size_t off = 0;
  auto alloc = [&](size_t bytes) -> void* {
    void* p = base + off; off = (off + bytes + 255) & ~(size_t)255; return p;
  };
  // bf16 weights, canonical [N][K]
  bf16* cW2  = (bf16*)alloc(256u * 256 * 2);
  bf16* cWQ  = (bf16*)alloc(256u * 512 * 2);
  bf16* cIHt = (bf16*)alloc(1536u * 768 * 2);
  bf16* cHHt = (bf16*)alloc(1536u * 512 * 2);
  bf16* cE2G = (bf16*)alloc(128u * 512 * 2);
  bf16* cIHg = (bf16*)alloc(384u * 128 * 2);
  bf16* cHHg = (bf16*)alloc(384u * 128 * 2);
  bf16* cG2T = (bf16*)alloc(512u * 128 * 2);
  bf16* cG2E = (bf16*)alloc(128u * 128 * 2);
  bf16* cG2D = (bf16*)alloc(128u * 128 * 2);
  bf16* cE2D = (bf16*)alloc(512u * 512 * 2);
  bf16* cPM  = (bf16*)alloc(32u * 512 * 2);
  bf16* cPL  = (bf16*)alloc(32u * 512 * 2);
  bf16* cC2D = (bf16*)alloc(512u * 544 * 2);
  bf16* cD2I = (bf16*)alloc(512u * 512 * 2);
  bf16* cIHd = (bf16*)alloc(1536u * 640 * 2);
  bf16* cHHd = (bf16*)alloc(1536u * 512 * 2);
  // activations
  float* TTC   = (float*)alloc(65536u * 4);
  float* DPV   = (float*)alloc(262144u * 4);
  float* STATS = (float*)alloc(256);
  bf16*  A1    = (bf16*)alloc((size_t)65536 * 256 * 2);
  bf16*  KM    = (bf16*)alloc((size_t)65536 * 256 * 2);
  float* NN    = (float*)alloc((size_t)65536 * 128 * 4);
  float* Q     = (float*)alloc(65536u * 4);
  float* CI    = (float*)alloc((size_t)NAG * 768 * 4);
  float* H     = (float*)alloc((size_t)NAG * HDIM * 4);
  float* GIE   = (float*)alloc((size_t)NAG * 1536 * 4);
  float* GHE   = (float*)alloc((size_t)NAG * 1536 * 4);
  float* GHID  = (float*)alloc((size_t)NAG * HQ * 4);
  float* GGH   = (float*)alloc((size_t)NAG * HQ * 4);
  float* GIG   = (float*)alloc((size_t)NAG * 384 * 4);
  float* GHG   = (float*)alloc((size_t)NAG * 384 * 4);
  float* GOALS = (float*)alloc((size_t)TDEC * NAG * HQ * 4);
  float* TGT   = (float*)alloc((size_t)TDEC * NAG * HDIM * 4);
  float* GFD   = (float*)alloc((size_t)TDEC * NAG * HQ * 4);
  float* GFES  = (float*)alloc((size_t)TDEC * NAG * HQ * 4);
  float* GFE   = (float*)alloc((size_t)NAG * HQ * 4);
  float* DEC0  = (float*)alloc((size_t)NAG * HDIM * 4);
  float* MU    = (float*)alloc((size_t)NAG * LAT * 4);
  float* LV    = (float*)alloc((size_t)NAG * LAT * 4);
  float* XCV   = (float*)alloc((size_t)NK * 544 * 4);
  float* DH    = (float*)alloc((size_t)NK * HDIM * 4);
  float* XDEC  = (float*)alloc((size_t)NK * 640 * 4);
  float* GID   = (float*)alloc((size_t)NK * 1536 * 4);
  float* GHD   = (float*)alloc((size_t)NK * 1536 * 4);

  auto cvt = [&](const float* src, bf16* dst, int Nr, int Kc, int tr) {
    int tot = Nr * Kc;
    k_wcvt<<<dim3((tot + 255) / 256), 256, 0, stream>>>(src, dst, Nr, Kc, tr);
  };
  auto gemm = [&](const void* A, int lda, const bf16* W, const float* bias,
                  void* C, int ldc, int M, int Nout, int Kin, int act,
                  int abf, int cbf) {
    dim3 g((M + 127) / 128, (Nout + 63) / 64);
    if (abf && cbf)
      gemm_wmma<1, 1><<<g, 256, 0, stream>>>(A, lda, W, bias, C, ldc, M, Nout, Kin, act);
    else
      gemm_wmma<0, 0><<<g, 256, 0, stream>>>(A, lda, W, bias, C, ldc, M, Nout, Kin, act);
  };
  auto fill = [&](float* p, int n, float v) {
    k_fill<<<dim3((n + 255) / 256), 256, 0, stream>>>(p, n, v);
  };
  auto copy = [&](float* d, const float* s, int n) {
    k_copy<<<dim3((n + 255) / 256), 256, 0, stream>>>(d, s, n);
  };

  // ---- weight conversion (linear weights (K,N) -> transpose; GRU already [N][K]) ----
  cvt(W_k2, cW2, 256, 256, 1);    cvt(W_q, cWQ, 256, 512, 1);
  cvt(Wih_t, cIHt, 1536, 768, 0); cvt(Whh_t, cHHt, 1536, 512, 0);
  cvt(W_e2g, cE2G, 128, 512, 1);
  cvt(Wih_g, cIHg, 384, 128, 0);  cvt(Whh_g, cHHg, 384, 128, 0);
  cvt(W_g2t, cG2T, 512, 128, 1);  cvt(W_g2e, cG2E, 128, 128, 1);
  cvt(W_g2d, cG2D, 128, 128, 1);  cvt(W_e2d, cE2D, 512, 512, 1);
  cvt(W_pm, cPM, 32, 512, 1);     cvt(W_pl, cPL, 32, 512, 1);
  cvt(W_c2d, cC2D, 512, 544, 1);  cvt(W_d2i, cD2I, 512, 512, 1);
  cvt(Wih_d, cIHd, 1536, 640, 0); cvt(Whh_d, cHHd, 1536, 512, 0);

  // ---- state init + constant probs output ----
  fill(H, NAG * HDIM, 0.f);
  fill(GFE, NAG * HQ, 0.f);
  fill(out2, NAG * TENC * KS, 1.f / (float)KS);

  for (int s = 0; s < TENC; ++s) {
    // ----- social encoder -----
    k_social_pairs<<<dim3(256), 256, 0, stream>>>(raw, s, TTC, DPV);
    k_stats<<<dim3(1), 256, 0, stream>>>(TTC, STATS);
    k_social_embed<<<dim3(NAG * NAG), 256, 0, stream>>>(
        TTC, DPV, STATS, W_k1, b_k1, W_nn, b_nn, A1, NN);
    gemm(A1, 256, cW2, b_k2, KM, 256, NAG * NAG, 256, 256, 1, 1, 1);   // k
    gemm(H, HDIM, cWQ, b_q, Q, 256, NAG, 256, HDIM, 1, 0, 0);          // q
    k_build_ci<<<dim3((NAG * 640 + 255) / 256), 256, 0, stream>>>(traj, s, GFE, CI);
    k_social_attn<<<dim3(NAG), 256, 0, stream>>>(KM, Q, NN, sw, CI);

    // ----- encoder GRU -----
    gemm(CI, 768, cIHt, bih_t, GIE, 1536, NAG, 1536, 768, 0, 0, 0);
    gemm(H, HDIM, cHHt, bhh_t, GHE, 1536, NAG, 1536, HDIM, 0, 0, 0);
    k_gru_gate<<<dim3((NAG * HDIM + 255) / 256), 256, 0, stream>>>(GIE, GHE, H, NAG, HDIM);

    // ----- SGE: goal GRU rollout -----
    gemm(H, HDIM, cE2G, b_e2g, GHID, HQ, NAG, HQ, HDIM, 1, 0, 0);
    copy(GGH, GHID, NAG * HQ);
    for (int t = 0; t < TDEC; ++t) {
      gemm(GGH, HQ, cIHg, bih_g, GIG, 384, NAG, 384, HQ, 0, 0, 0);
      gemm(GGH, HQ, cHHg, bhh_g, GHG, 384, NAG, 384, HQ, 0, 0, 0);
      k_gru_gate<<<dim3((NAG * HQ + 255) / 256), 256, 0, stream>>>(GIG, GHG, GGH, NAG, HQ);
      copy(GOALS + (size_t)t * NAG * HQ, GGH, NAG * HQ);
    }
    gemm(GOALS, HQ, cG2T, b_g2t, TGT, HDIM, TDEC * NAG, HDIM, HQ, 1, 0, 0);
    k_reg_goal<<<dim3((TDEC * NAG * 2 + 255) / 256), 256, 0, stream>>>(
        TGT, W_reg, b_reg, s, out0);
    gemm(GOALS, HQ, cG2D, b_g2d, GFD, HQ, TDEC * NAG, HQ, HQ, 1, 0, 0);
    gemm(GOALS, HQ, cG2E, b_g2e, GFES, HQ, TDEC * NAG, HQ, HQ, 1, 0, 0);
    k_enc_attn<<<dim3(NAG), 128, 0, stream>>>(GFES, W_ea, b_ea, GFE);

    // ----- CVAE prior + decoder init -----
    gemm(H, HDIM, cE2D, b_e2d, DEC0, HDIM, NAG, HDIM, HDIM, 1, 0, 0);
    gemm(DEC0, HDIM, cPM, b_pm, MU, LAT, NAG, LAT, HDIM, 0, 0, 0);
    gemm(DEC0, HDIM, cPL, b_pl, LV, LAT, NAG, LAT, HDIM, 0, 0, 0);
    k_build_xcvae<<<dim3((NK * 544 + 255) / 256), 256, 0, stream>>>(
        DEC0, MU, LV, eps, s, XCV);
    gemm(XCV, 544, cC2D, b_c2d, DH, HDIM, NK, HDIM, 544, 1, 0, 0);

    // ----- CVAE decoder rollout -----
    for (int t = 0; t < TDEC; ++t) {
      k_dec_attn<<<dim3(NAG), 128, 0, stream>>>(GFD, W_da, b_da, t, XDEC);
      gemm(DH, HDIM, cD2I, b_d2i, XDEC + 128, 640, NK, HDIM, HDIM, 1, 0, 0); // ddi
      gemm(XDEC, 640, cIHd, bih_d, GID, 1536, NK, 1536, 640, 0, 0, 0);
      gemm(DH, HDIM, cHHd, bhh_d, GHD, 1536, NK, 1536, HDIM, 0, 0, 0);
      k_gru_gate<<<dim3((NK * HDIM + 255) / 256), 256, 0, stream>>>(GID, GHD, DH, NK, HDIM);
      k_reg_dec<<<dim3((NK * 2 + 255) / 256), 256, 0, stream>>>(
          DH, W_reg, b_reg, s, t, out1);
    }
  }
}
